// FourierKAN_54279796687467
// MI455X (gfx1250) — matrix operand order repaired and checked by reading, not compiled, not run on hardware
//
#include <hip/hip_runtime.h>
#include <hip/hip_bf16.h>
#include <stdint.h>

// MI455X / gfx1250, wave32. Compute-bound (~87 GFLOP vs ~50MB L2-resident data):
// f16 WMMA (v_wmma_f32_16x16x32_f16), f32 accumulate. Coefficients pre-packed to
// f16 tiles once per launch; hot loop streams them to LDS with
// global_load_async_to_lds_b128 (ASYNCcnt), 3-deep ring, 1 barrier per K-step.

typedef __attribute__((ext_vector_type(16))) _Float16     v16h;
typedef __attribute__((ext_vector_type(8)))  float        v8f;
typedef __attribute__((ext_vector_type(8)))  unsigned int v8u;

#define BM 64
#define BN 128
#define AS_STRIDE 20   // 16 used + 4 pad; 80B row: 16B-aligned, conflict-free banks
#define BS_STRIDE 20

static __device__ __forceinline__ uint32_t packf16(float a, float b) {
    // v_cvt_pk_rtz_f16_f32; builtin returns __fp16 ext_vector(2) -> bit-cast blind
    return __builtin_bit_cast(uint32_t, __builtin_amdgcn_cvt_pkrtz(a, b));
}

static __device__ __forceinline__ v16h frag_cat(uint4 lo, uint4 hi) {
    v8u u;
    u[0] = lo.x; u[1] = lo.y; u[2] = lo.z; u[3] = lo.w;
    u[4] = hi.x; u[5] = hi.y; u[6] = hi.z; u[7] = hi.w;
    return __builtin_bit_cast(v16h, u);
}

// Per-lane async copy of 16B global -> LDS (tracked by ASYNCcnt).
// VDST = VGPR with LDS byte address (low 32 bits of generic shared pointer).
static __device__ __forceinline__ void async_b128(uint32_t lds_addr, const void* gaddr) {
    asm volatile("global_load_async_to_lds_b128 %0, %1, off"
                 :: "v"(lds_addr), "v"((unsigned long long)(uintptr_t)gaddr)
                 : "memory");
}
static __device__ __forceinline__ void wait_async_le2() {
    asm volatile("s_wait_asynccnt 0x2" ::: "memory");
}
static __device__ __forceinline__ void wait_async_0() {
    asm volatile("s_wait_asynccnt 0x0" ::: "memory");
}

// A layout (16-bit 16x32): lanes 0-15 hold K0..15, lanes 16-31
// K16..31. B (32x16): lanes 0-15 K0..15, lanes 16-31 K16..31, col=lane&15.
// As/Bs store dword j = K/2 per row/col.
static __device__ __forceinline__ void frags_mma(
    const uint32_t (*As)[AS_STRIDE], const uint32_t (*Bs)[BS_STRIDE],
    int waveM, int waveN, int half, int m, v8f acc[2][2])
{
    v16h a[2], b[2];
#pragma unroll
    for (int mf = 0; mf < 2; ++mf) {
        const uint32_t* ap = &As[waveM * 32 + mf * 16 + m][0];
        uint4 lo = *(const uint4*)(ap + half * 8);
        uint4 hi = *(const uint4*)(ap + half * 8 + 4);
        a[mf] = frag_cat(lo, hi);
    }
#pragma unroll
    for (int nf = 0; nf < 2; ++nf) {
        const uint32_t* bp = &Bs[waveN * 32 + nf * 16 + m][0];
        uint4 lo = *(const uint4*)(bp + half * 8);
        uint4 hi = *(const uint4*)(bp + half * 8 + 4);
        b[nf] = frag_cat(lo, hi);
    }
#pragma unroll
    for (int mf = 0; mf < 2; ++mf)
#pragma unroll
        for (int nf = 0; nf < 2; ++nf)
            acc[mf][nf] = __builtin_amdgcn_wmma_f32_16x16x32_f16(
                false, a[mf], false, b[nf], (short)0, acc[mf][nf], false, false);
}

// ---- One-time coeff pre-pack: f32 (2,OUT,IN,GRID) -> f16 pairs laid out as
// Wpk[colBlock][step][128][16] (dword j = K/2 of the 32-K step tile).
template<int IN, int OUT, int GRID>
__global__ __launch_bounds__(256) void pack_coeffs(
    const float* __restrict__ coeffs, uint32_t* __restrict__ Wpk)
{
    constexpr int FPS = 32 / (2 * GRID);
    constexpr int NSTEPS = IN / FPS;
    const int idx = blockIdx.x * 256 + threadIdx.x;   // one packed dword each
    const int j    = idx & 15;
    const int ol   = (idx >> 4) & 127;
    const int step = (idx >> 11) % NSTEPS;
    const int cb   = idx / (2048 * NSTEPS);
    const int o    = cb * 128 + ol;
    const int kk   = j * 2;
    const int f    = kk / (2 * GRID);
    const int r    = kk % (2 * GRID);
    const int c    = r / GRID;
    const int g    = r % GRID;   // even; g+1 stays in same (c,f) region
    const float* p = coeffs + (((size_t)c * OUT + o) * IN + (step * FPS + f)) * GRID + g;
    Wpk[idx] = packf16(p[0], p[1]);
}

// ---------------- Layer 1: H = x (4096x256) * W1^T + b1 ------------------------
// Also emits deterministic per-block (sum, sumsq) partials for BatchNorm stats.
__global__ __launch_bounds__(256) void gemm1_kernel(
    const float* __restrict__ X, const float* __restrict__ W1,
    const float* __restrict__ b1, float* __restrict__ H,
    float* __restrict__ partials)
{
    __shared__ __align__(16) uint32_t As[BM][AS_STRIDE];
    __shared__ __align__(16) uint32_t Bs[BN][BS_STRIDE];
    __shared__ float redS[256];
    __shared__ float redQ[256];

    const int tid = threadIdx.x;
    const int rowBase = blockIdx.x * BM;
    const int colBase = blockIdx.y * BN;
    const int lane = tid & 31, wave = tid >> 5;
    const int waveM = wave >> 2, waveN = wave & 3;
    const int half = lane >> 4, m = lane & 15;

    const int arow = tid >> 2, aq = tid & 3;
    const int bcol = tid >> 1, bkh = tid & 1;

    v8f acc[2][2] = {};

    for (int k0 = 0; k0 < 256; k0 += 32) {
        __syncthreads();
        {
            const float* xp = X + (size_t)(rowBase + arow) * 256 + k0 + aq * 8;
            float4 w0 = *(const float4*)xp;
            float4 w1 = *(const float4*)(xp + 4);
            As[arow][aq * 4 + 0] = packf16(w0.x, w0.y);
            As[arow][aq * 4 + 1] = packf16(w0.z, w0.w);
            As[arow][aq * 4 + 2] = packf16(w1.x, w1.y);
            As[arow][aq * 4 + 3] = packf16(w1.z, w1.w);
        }
        {
            const float* wp = W1 + (size_t)(colBase + bcol) * 256 + k0 + bkh * 16;
#pragma unroll
            for (int ch = 0; ch < 4; ++ch) {
                float4 w = *(const float4*)(wp + ch * 4);
                Bs[bcol][bkh * 8 + ch * 2 + 0] = packf16(w.x, w.y);
                Bs[bcol][bkh * 8 + ch * 2 + 1] = packf16(w.z, w.w);
            }
        }
        __syncthreads();
        frags_mma(As, Bs, waveM, waveN, half, m, acc);
    }

    float lsum = 0.f, lsq = 0.f;
#pragma unroll
    for (int mf = 0; mf < 2; ++mf)
#pragma unroll
        for (int nf = 0; nf < 2; ++nf) {
            const int col = colBase + waveN * 32 + nf * 16 + m;
            const float bia = b1[col];
#pragma unroll
            for (int r = 0; r < 8; ++r) {
                const int row = rowBase + waveM * 32 + mf * 16 + half * 8 + r;
                float v = acc[mf][nf][r] + bia;
                H[(size_t)row * 512 + col] = v;
                lsum += v; lsq += v * v;
            }
        }
    redS[tid] = lsum; redQ[tid] = lsq;
    __syncthreads();
    for (int s = 128; s > 0; s >>= 1) {
        if (tid < s) { redS[tid] += redS[tid + s]; redQ[tid] += redQ[tid + s]; }
        __syncthreads();
    }
    if (tid == 0) {
        const int bid = blockIdx.y * gridDim.x + blockIdx.x;
        partials[2 * bid]     = redS[0];
        partials[2 * bid + 1] = redQ[0];
    }
}

__global__ void stats_finalize(const float* __restrict__ partials,
                               float* __restrict__ stats)
{
    if (threadIdx.x == 0 && blockIdx.x == 0) {
        float s = 0.f, s2 = 0.f;
        for (int i = 0; i < 256; ++i) { s += partials[2 * i]; s2 += partials[2 * i + 1]; }
        const float N = 4096.0f * 512.0f;
        const float mean = s / N;
        const float var = s2 / N - mean * mean;
        stats[0] = mean;
        stats[1] = rsqrtf(var + 1e-5f);
    }
}

// ---------------- FourierKAN layer: basis-on-the-fly f16 WMMA GEMM -------------
// K = IN*2*GRID. One 32-K step per FPS input features. B tiles arrive via async
// DMA (3-buffer ring); A basis generated with v_cos/v_sin into a double buffer.
// One barrier per step. Layer-2 fuses BatchNorm+ReLU into the Xs staging.
template<int IN, int OUT, int GRID, bool NORM>
__global__ __launch_bounds__(256) void kan_kernel(
    const float* __restrict__ X,        // (4096 x IN)
    const uint32_t* __restrict__ Wpk,   // pre-packed f16 coeff tiles
    const float* __restrict__ bias,     // (OUT)
    const float* __restrict__ stats,    // mean, invstd
    const float* __restrict__ gamma,
    const float* __restrict__ beta,
    float* __restrict__ Y)              // (4096 x OUT)
{
    constexpr int FPS = 32 / (2 * GRID);
    constexpr int CF  = 8;
    constexpr int SPC = CF / FPS;
    constexpr int NSTEPS = IN / FPS;

    __shared__ __align__(16) uint32_t As[2][BM][AS_STRIDE];
    __shared__ __align__(16) uint32_t Bs[3][BN][BS_STRIDE];
    __shared__ float Xs[BM][CF];

    const int tid = threadIdx.x;
    const int rowBase = blockIdx.x * BM;
    const int cb = blockIdx.y;
    const int colBase = cb * BN;
    const int lane = tid & 31, wave = tid >> 5;
    const int waveM = wave >> 2, waveN = wave & 3;
    const int half = lane >> 4, m = lane & 15;

    float gmul = 1.f, gadd = 0.f;
    if constexpr (NORM) {
        const float mean = stats[0], inv = stats[1];
        gmul = gamma[0] * inv;
        gadd = beta[0] - gmul * mean;
    }

    // basis role
    const int arow = tid >> 2, aq = tid & 3;
    const int aK0  = aq * 8;
    const int aF   = aK0 / (2 * GRID);
    const int aR   = aK0 % (2 * GRID);
    const bool aSin = (aR >= GRID);
    const int aG0  = aR % GRID;

    // B DMA role: 2 lanes per output col, 2 x b128 each per step
    const int bcol = tid >> 1, bkh = tid & 1;
    const uint32_t* wbase = Wpk + (size_t)cb * NSTEPS * 2048 + bcol * 16 + bkh * 8;

    v8f acc[2][2] = {};

    // prologue: DMA step-0 B tile into ring slot 0
    {
        const uint32_t lds0 = (uint32_t)(uintptr_t)&Bs[0][bcol][bkh * 8];
        async_b128(lds0,      wbase);
        async_b128(lds0 + 16, wbase + 4);
    }

    int sg = 0;
    for (int i0 = 0; i0 < IN; i0 += CF) {
#pragma unroll
        for (int it = 0; it < (BM * CF) / 256; ++it) {
            const int idx = it * 256 + tid;
            const int r = idx >> 3, fc = idx & 7;
            float xv = X[(size_t)(rowBase + r) * IN + i0 + fc];
            if constexpr (NORM) xv = fmaxf(fmaf(gmul, xv, gadd), 0.f);
            Xs[r][fc] = xv;
        }
        __syncthreads();
        for (int s = 0; s < SPC; ++s, ++sg) {
            const int ab = sg & 1;
            const int rb = sg % 3;
            { // basis -> As[ab] (overlaps with in-flight B DMA)
                const float xv = Xs[arow][s * FPS + aF];
                float v[8];
#pragma unroll
                for (int e = 0; e < 8; ++e) {
                    const float ang = (float)(aG0 + e + 1) * xv;
                    v[e] = aSin ? __sinf(ang) : __cosf(ang);
                }
#pragma unroll
                for (int j = 0; j < 4; ++j)
                    As[ab][arow][aq * 4 + j] = packf16(v[2 * j], v[2 * j + 1]);
            }
            if (sg + 1 < NSTEPS) { // DMA next step's B tile into ring slot sg+1
                const int nb = (sg + 1) % 3;
                const uint32_t lds0 = (uint32_t)(uintptr_t)&Bs[nb][bcol][bkh * 8];
                const uint32_t* gp = wbase + (size_t)(sg + 1) * 2048;
                async_b128(lds0,      gp);
                async_b128(lds0 + 16, gp + 4);
                wait_async_le2();   // current tile landed; next stays in flight
            } else {
                wait_async_0();
            }
            __syncthreads();
            frags_mma(As[ab], Bs[rb], waveM, waveN, half, m, acc);
        }
    }

#pragma unroll
    for (int mf = 0; mf < 2; ++mf)
#pragma unroll
        for (int nf = 0; nf < 2; ++nf) {
            const int col = colBase + waveN * 32 + nf * 16 + m;
            const float bia = bias[col];
#pragma unroll
            for (int r = 0; r < 8; ++r) {
                const int row = rowBase + waveM * 32 + mf * 16 + half * 8 + r;
                Y[(size_t)row * OUT + col] = acc[mf][nf][r] + bia;
            }
        }
}

// ---------------- Softmax over 256 logits per row ------------------------------
__global__ __launch_bounds__(256) void softmax256(const float* __restrict__ Y,
                                                  float* __restrict__ out)
{
    __shared__ float red[256];
    const int row = blockIdx.x;
    const int c = threadIdx.x;
    const float v = Y[(size_t)row * 256 + c];
    red[c] = v;
    __syncthreads();
    for (int s = 128; s > 0; s >>= 1) {
        if (c < s) red[c] = fmaxf(red[c], red[c + s]);
        __syncthreads();
    }
    const float mx = red[0];
    __syncthreads();
    const float e = __expf(v - mx);
    red[c] = e;
    __syncthreads();
    for (int s = 128; s > 0; s >>= 1) {
        if (c < s) red[c] += red[c + s];
        __syncthreads();
    }
    out[(size_t)row * 256 + c] = e / red[0];
}

extern "C" void kernel_launch(void* const* d_in, const int* in_sizes, int n_in,
                              void* d_out, int out_size, void* d_ws, size_t ws_size,
                              hipStream_t stream)
{
    (void)in_sizes; (void)n_in; (void)out_size; (void)ws_size;

    const float* x       = (const float*)d_in[0]; // (4096,1,256)
    const float* W1      = (const float*)d_in[1]; // (512,256)
    const float* b1      = (const float*)d_in[2]; // (512)
    const float* gamma   = (const float*)d_in[3]; // (1)
    const float* beta    = (const float*)d_in[4]; // (1)
    const float* coeffs2 = (const float*)d_in[5]; // (2,512,512,16)
    const float* bias2   = (const float*)d_in[6]; // (512)
    const float* coeffs3 = (const float*)d_in[7]; // (2,256,512,8)
    const float* bias3   = (const float*)d_in[8]; // (256)
    float* out = (float*)d_out;                   // (4096,256)

    // Workspace byte layout (~42 MB total).
    char* ws = (char*)d_ws;
    float*    H        = (float*)(ws);                     //  8 MB
    float*    H2       = (float*)(ws + 8388608);           //  8 MB
    float*    Y3       = (float*)(ws + 16777216);          //  4 MB
    float*    partials = (float*)(ws + 20971520);          //  2 KB (rewritten each call)
    float*    stats    = (float*)(ws + 20975616);          //  8 B
    uint32_t* Wpk2     = (uint32_t*)(ws + 20979712);       // 16 MB packed coeffs2
    uint32_t* Wpk3     = (uint32_t*)(ws + 37756928);       //  4 MB packed coeffs3

    pack_coeffs<512, 512, 16><<<4194304 / 256, 256, 0, stream>>>(coeffs2, Wpk2);
    pack_coeffs<512, 256,  8><<<1048576 / 256, 256, 0, stream>>>(coeffs3, Wpk3);
    gemm1_kernel<<<dim3(64, 4), 256, 0, stream>>>(x, W1, b1, H, partials);
    stats_finalize<<<1, 1, 0, stream>>>(partials, stats);
    kan_kernel<512, 512, 16, true ><<<dim3(64, 4), 256, 0, stream>>>(
        H,  Wpk2, bias2, stats, gamma, beta, H2);
    kan_kernel<512, 256,  8, false><<<dim3(64, 2), 256, 0, stream>>>(
        H2, Wpk3, bias3, stats, gamma, beta, Y3);
    softmax256<<<4096, 256, 0, stream>>>(Y3, out);
}
